// CausalSelfAttention_22454089024077
// MI455X (gfx1250) — matrix operand order repaired and checked by reading, not compiled, not run on hardware
//
#include <hip/hip_runtime.h>
#include <hip/hip_bf16.h>

// Problem constants (match reference)
#define B_  2
#define T_  2048
#define C_  1024
#define H_  16
#define DH_ 64          // head dim
#define HD2 32          // Dh/2

typedef __attribute__((ext_vector_type(16))) __bf16         v16bf;
typedef __attribute__((ext_vector_type(8)))  float          v8f;
typedef __attribute__((ext_vector_type(8)))  unsigned short u16x8;

union BF16Frag { u16x8 h[2]; v16bf v; };

__device__ __forceinline__ unsigned short f2bf(float x) {
    unsigned int u = __builtin_bit_cast(unsigned int, x);
    u += 0x7FFFu + ((u >> 16) & 1u);      // round-to-nearest-even
    return (unsigned short)(u >> 16);
}

__device__ __forceinline__ v16bf load_frag(const unsigned short* p0,
                                           const unsigned short* p1) {
    BF16Frag f;
    f.h[0] = *(const u16x8*)p0;
    f.h[1] = *(const u16x8*)p1;
    return f.v;
}

// ---------------------------------------------------------------------------
// Tensor Data Mover (gfx1250): 2D tile load Global -> LDS, bf16 elements.
#if __has_builtin(__builtin_amdgcn_tensor_load_to_lds) && \
    __has_builtin(__builtin_amdgcn_s_wait_tensorcnt)
#define USE_TDM 1
#else
#define USE_TDM 0
#endif

#if USE_TDM
typedef __attribute__((ext_vector_type(4))) unsigned int u32x4;
typedef __attribute__((ext_vector_type(8))) int          i32x8;
typedef __attribute__((ext_vector_type(4))) int          i32x4;

// D# packing per CDNA5 ISA ch.8 (2D tile, 2-byte elements, no pad/iterate).
__device__ __forceinline__ void tdm_load_2d(unsigned int lds_off,
                                            const void* gtile,
                                            unsigned int tile_d0,   // contiguous elems/row
                                            unsigned int tile_d1,   // rows
                                            unsigned int tens_d0,   // tensor row length
                                            unsigned int tens_d1,   // tensor rows
                                            unsigned long long stride0) { // elems between rows
    unsigned long long ga = (unsigned long long)(uintptr_t)gtile;
    u32x4 g0;
    g0[0] = 1u;                                       // count=1, user descriptor
    g0[1] = lds_off;                                  // LDS byte address
    g0[2] = (unsigned int)(ga & 0xFFFFFFFFu);         // global_addr[31:0]
    g0[3] = (unsigned int)((ga >> 32) & 0x01FFFFFFu)  // global_addr[56:32]
          | (2u << 30);                               // type = 2 ("image")
    i32x8 g1 = {};
    g1[0] = (int)(1u << 16);                          // data_size=1 (2B), wg_mask=0
    g1[1] = (int)((tens_d0 & 0xFFFFu) << 16);                          // tensor_dim0 lo
    g1[2] = (int)(((tens_d0 >> 16) & 0xFFFFu) | ((tens_d1 & 0xFFFFu) << 16));
    g1[3] = (int)(((tens_d1 >> 16) & 0xFFFFu) | ((tile_d0 & 0xFFFFu) << 16));
    g1[4] = (int)(tile_d1 & 0xFFFFu);                 // tile_dim1; tile_dim2=0
    g1[5] = (int)(unsigned int)(stride0 & 0xFFFFFFFFull);              // dim0_stride lo
    g1[6] = (int)(unsigned int)((stride0 >> 32) & 0xFFFFull);          // dim0_stride hi
    g1[7] = 0;
    i32x4 z4 = {0, 0, 0, 0};
    i32x8 z8 = {};
    __builtin_amdgcn_tensor_load_to_lds(g0, g1, z4, z4, z8, 0);
}
#endif

// ---------------------------------------------------------------------------
// f32 -> bf16 elementwise convert
__global__ void cvt_bf16_kernel(const float* __restrict__ in,
                                unsigned short* __restrict__ out, int n) {
    int i = blockIdx.x * blockDim.x + threadIdx.x;
    if (i < n) out[i] = f2bf(in[i]);
}

// f32 [K, N] row-major  ->  bf16 [N, K] row-major (transposed weight pack)
__global__ void transpose_bf16_kernel(const float* __restrict__ in,
                                      unsigned short* __restrict__ out,
                                      int K, int N) {
    int idx = blockIdx.x * blockDim.x + threadIdx.x;
    if (idx >= K * N) return;
    int n = idx % N;
    int k = idx / N;
    out[(size_t)n * K + k] = f2bf(in[(size_t)k * N + n]);
}

// ---------------------------------------------------------------------------
// GEMM: D[M,N] (f32) = A[M,K] (bf16, row-major) x BT[N,K] (bf16, row-major = B^T)
// One wave computes a 32x32 tile (2 A-frags x 2 B-frags -> 4 WMMA / k-step).
// Block = 256 threads = 8 waves arranged 2(M) x 4(N): block tile 64M x 128N.
__global__ void gemm_bf16_wmma(const unsigned short* __restrict__ A,
                               const unsigned short* __restrict__ BT,
                               float* __restrict__ D,
                               int M, int N, int K) {
    const int lane = threadIdx.x & 31;
    const int wave = threadIdx.x >> 5;
    const int wm   = wave >> 2;          // 0..1
    const int wn   = wave & 3;           // 0..3
    const int m0   = blockIdx.y * 64 + wm * 32;
    const int n0   = blockIdx.x * 128 + wn * 32;
    const int col  = lane & 15;
    const bool hi  = lane >= 16;

    const unsigned short* arow0 = A  + (size_t)(m0 + col) * K;
    const unsigned short* arow1 = arow0 + (size_t)16 * K;
    const unsigned short* brow0 = BT + (size_t)(n0 + col) * K;
    const unsigned short* brow1 = brow0 + (size_t)16 * K;
    const int aoff = hi ? 16 : 0;        // A-fragment K sub-offset
    const int boff = hi ? 16 : 0;        // B-fragment K sub-offset

    v8f acc00 = {}, acc01 = {}, acc10 = {}, acc11 = {};
    for (int k = 0; k < K; k += 32) {
        __builtin_prefetch(arow0 + k + 256, 0, 1);
        __builtin_prefetch(brow0 + k + 256, 0, 1);
        v16bf a0 = load_frag(arow0 + k + aoff, arow0 + k + aoff + 8);
        v16bf a1 = load_frag(arow1 + k + aoff, arow1 + k + aoff + 8);
        v16bf b0 = load_frag(brow0 + k + boff, brow0 + k + boff + 8);
        v16bf b1 = load_frag(brow1 + k + boff, brow1 + k + boff + 8);
        acc00 = __builtin_amdgcn_wmma_f32_16x16x32_bf16(
                    false, a0, false, b0, (short)0, acc00, false, false);
        acc01 = __builtin_amdgcn_wmma_f32_16x16x32_bf16(
                    false, a0, false, b1, (short)0, acc01, false, false);
        acc10 = __builtin_amdgcn_wmma_f32_16x16x32_bf16(
                    false, a1, false, b0, (short)0, acc10, false, false);
        acc11 = __builtin_amdgcn_wmma_f32_16x16x32_bf16(
                    false, a1, false, b1, (short)0, acc11, false, false);
    }

    const int r0 = m0 + (hi ? 8 : 0);
#pragma unroll
    for (int r = 0; r < 8; ++r) {
        float* d0 = D + (size_t)(r0 + r) * N + n0 + col;
        float* d1 = D + (size_t)(r0 + 16 + r) * N + n0 + col;
        d0[0]  = acc00[r];
        d0[16] = acc01[r];
        d1[0]  = acc10[r];
        d1[16] = acc11[r];
    }
}

// ---------------------------------------------------------------------------
// RoPE + re-layout:  qkv f32 [B,T,3C]  ->
//   Q bf16 [B,H,T,Dh], K bf16 [B,H,T,Dh], VT bf16 [B,H,Dh,T]
__global__ void rope_scatter_kernel(const float* __restrict__ qkv,
                                    const float* __restrict__ fcos,
                                    const float* __restrict__ fsin,
                                    unsigned short* __restrict__ Qb,
                                    unsigned short* __restrict__ Kb,
                                    unsigned short* __restrict__ VTb) {
    int idx = blockIdx.x * blockDim.x + threadIdx.x;   // (b,t,h,i)
    if (idx >= B_ * T_ * H_ * HD2) return;
    int i = idx & (HD2 - 1);
    int h = (idx >> 5) & (H_ - 1);
    int t = (idx >> 9) & (T_ - 1);
    int b = idx >> 20;

    const float* row = qkv + (size_t)(b * T_ + t) * (3 * C_);
    int off = h * DH_ + 2 * i;
    float c = fcos[t * HD2 + i];
    float s = fsin[t * HD2 + i];

    float qa = row[off],          qb = row[off + 1];
    float ka = row[C_ + off],     kb = row[C_ + off + 1];
    float va = row[2 * C_ + off], vb = row[2 * C_ + off + 1];

    size_t qi = ((size_t)(b * H_ + h) * T_ + t) * DH_ + 2 * i;
    Qb[qi]     = f2bf(qa * c - qb * s);
    Qb[qi + 1] = f2bf(qa * s + qb * c);
    Kb[qi]     = f2bf(ka * c - kb * s);
    Kb[qi + 1] = f2bf(ka * s + kb * c);

    size_t vi = ((size_t)(b * H_ + h) * DH_ + 2 * i) * T_ + t;
    VTb[vi]      = f2bf(va);
    VTb[vi + T_] = f2bf(vb);
}

// ---------------------------------------------------------------------------
// Flash attention: one wave per (b, h, 16-query tile). Online softmax in LDS.
// K/V tiles are staged into LDS with the Tensor Data Mover when available.
__global__ void flash_attn_kernel(const unsigned short* __restrict__ Q,
                                  const unsigned short* __restrict__ Kb,
                                  const unsigned short* __restrict__ VT,
                                  unsigned short* __restrict__ Y) {
    const int lane  = threadIdx.x;
    const int q0    = blockIdx.x * 16;
    const int h     = blockIdx.y;
    const int b     = blockIdx.z;
    const int col   = lane & 15;
    const bool hiL  = lane >= 16;
    const int aoff  = hiL ? 16 : 0;
    const int rbase = hiL ? 8 : 0;

    __shared__ __align__(16) float          sS[16][32];
    __shared__ float sM[16], sL[16], sF[16];
    __shared__ __align__(16) unsigned short sP[16][32];
#if USE_TDM
    __shared__ __align__(16) unsigned short Kt[32 * 64];  // [key_local][d]
    __shared__ __align__(16) unsigned short Vt[64 * 32];  // [d][key_local]
#endif

    if (lane < 16) { sM[lane] = -3.0e38f; sL[lane] = 0.0f; }
    __syncthreads();

    const size_t head  = (size_t)(b * H_ + h) * T_ * DH_;
    const unsigned short* qrow  = Q + head + (size_t)(q0 + col) * DH_;
    const unsigned short* vbase = VT + (size_t)(b * H_ + h) * DH_ * T_;
    v16bf qf0 = load_frag(qrow + aoff,      qrow + aoff + 8);
    v16bf qf1 = load_frag(qrow + 32 + aoff, qrow + 32 + aoff + 8);

    v8f acc0 = {}, acc1 = {}, acc2 = {}, acc3 = {};
    const float scale = 0.125f;   // 1/sqrt(64)
    const int kblocks = ((q0 + 15) >> 5);

    for (int kb = 0; kb <= kblocks; ++kb) {
        const int k0 = kb * 32;
        const int boffB = hiL ? 16 : 0;

#if USE_TDM
        // Stage K tile (32 keys x 64 dims) and V tile (64 dims x 32 keys).
        tdm_load_2d((unsigned int)(uintptr_t)Kt,
                    Kb + head + (size_t)k0 * DH_,
                    /*tile*/ DH_, 32, /*tensor*/ DH_, T_, /*stride*/ DH_);
        tdm_load_2d((unsigned int)(uintptr_t)Vt,
                    vbase + k0,
                    /*tile*/ 32, DH_, /*tensor*/ T_, DH_, /*stride*/ T_);
        __builtin_amdgcn_s_wait_tensorcnt(0);
#endif

        // ---- scores: S[16q x 32k] via two 16x16 WMMA tiles (K-dim = Dh) ----
        v8f s[2];
#pragma unroll
        for (int t = 0; t < 2; ++t) {
#if USE_TDM
            const unsigned short* krow = &Kt[(t * 16 + col) * DH_];
#else
            const unsigned short* krow =
                Kb + head + (size_t)(k0 + t * 16 + col) * DH_;
#endif
            v16bf kf0 = load_frag(krow + boffB,      krow + boffB + 8);
            v16bf kf1 = load_frag(krow + 32 + boffB, krow + 32 + boffB + 8);
            v8f st = {};
            st = __builtin_amdgcn_wmma_f32_16x16x32_bf16(
                     false, qf0, false, kf0, (short)0, st, false, false);
            st = __builtin_amdgcn_wmma_f32_16x16x32_bf16(
                     false, qf1, false, kf1, (short)0, st, false, false);
            s[t] = st;
        }

        // ---- mask + scale, stage into LDS (C-layout -> row-major) ----
#pragma unroll
        for (int t = 0; t < 2; ++t) {
#pragma unroll
            for (int r = 0; r < 8; ++r) {
                int m   = rbase + r;
                int key = k0 + t * 16 + col;
                float v = s[t][r] * scale;
                if (key > q0 + m) v = -1.0e30f;
                sS[m][t * 16 + col] = v;
            }
        }
        __syncthreads();

        // ---- online softmax per query row (lanes 0..15 own one row each) --
        if (lane < 16) {
            int m = lane;
            float oldM = sM[m], mx = oldM;
#pragma unroll
            for (int k2 = 0; k2 < 32; ++k2) mx = fmaxf(mx, sS[m][k2]);
            float f   = __expf(oldM - mx);
            float sum = sL[m] * f;
#pragma unroll
            for (int k2 = 0; k2 < 32; ++k2) {
                float p = __expf(sS[m][k2] - mx);
                sum += p;
                sP[m][k2] = f2bf(p);
            }
            sM[m] = mx; sL[m] = sum; sF[m] = f;
        }
        __syncthreads();

        // ---- rescale running output accumulators ----
#pragma unroll
        for (int r = 0; r < 8; ++r) {
            float f = sF[rbase + r];
            acc0[r] *= f; acc1[r] *= f; acc2[r] *= f; acc3[r] *= f;
        }

        // ---- O += P (16x32, A-layout from LDS)  x  V (32x16 per Dh tile) --
        v16bf pf = load_frag(&sP[col][aoff], &sP[col][aoff + 8]);

#if USE_TDM
        const unsigned short* vr0 = &Vt[(0  + col) * 32 + boffB];
        const unsigned short* vr1 = &Vt[(16 + col) * 32 + boffB];
        const unsigned short* vr2 = &Vt[(32 + col) * 32 + boffB];
        const unsigned short* vr3 = &Vt[(48 + col) * 32 + boffB];
#else
        const unsigned short* vr0 = vbase + (size_t)(0  + col) * T_ + k0 + boffB;
        const unsigned short* vr1 = vbase + (size_t)(16 + col) * T_ + k0 + boffB;
        const unsigned short* vr2 = vbase + (size_t)(32 + col) * T_ + k0 + boffB;
        const unsigned short* vr3 = vbase + (size_t)(48 + col) * T_ + k0 + boffB;
#endif
        v16bf vf0 = load_frag(vr0, vr0 + 8);
        v16bf vf1 = load_frag(vr1, vr1 + 8);
        v16bf vf2 = load_frag(vr2, vr2 + 8);
        v16bf vf3 = load_frag(vr3, vr3 + 8);

        acc0 = __builtin_amdgcn_wmma_f32_16x16x32_bf16(
                   false, pf, false, vf0, (short)0, acc0, false, false);
        acc1 = __builtin_amdgcn_wmma_f32_16x16x32_bf16(
                   false, pf, false, vf1, (short)0, acc1, false, false);
        acc2 = __builtin_amdgcn_wmma_f32_16x16x32_bf16(
                   false, pf, false, vf2, (short)0, acc2, false, false);
        acc3 = __builtin_amdgcn_wmma_f32_16x16x32_bf16(
                   false, pf, false, vf3, (short)0, acc3, false, false);

        __syncthreads();   // protect sS/sP/Kt/Vt before next block overwrites
    }

    // ---- epilogue: divide by softmax denominator, write bf16 [B,T,C] ----
#pragma unroll
    for (int r = 0; r < 8; ++r) {
        int m = rbase + r;
        float inv = 1.0f / sL[m];
        size_t orow = ((size_t)b * T_ + q0 + m) * C_ + h * DH_;
        Y[orow + 0  + col] = f2bf(acc0[r] * inv);
        Y[orow + 16 + col] = f2bf(acc1[r] * inv);
        Y[orow + 32 + col] = f2bf(acc2[r] * inv);
        Y[orow + 48 + col] = f2bf(acc3[r] * inv);
    }
}

// ---------------------------------------------------------------------------
extern "C" void kernel_launch(void* const* d_in, const int* in_sizes, int n_in,
                              void* d_out, int out_size, void* d_ws, size_t ws_size,
                              hipStream_t stream) {
    const float* x      = (const float*)d_in[0];   // [B,T,C]
    const float* w_attn = (const float*)d_in[1];   // [C,3C]
    const float* w_proj = (const float*)d_in[2];   // [C,C]
    const float* fcos   = (const float*)d_in[3];   // [T,Dh/2]
    const float* fsin   = (const float*)d_in[4];   // [T,Dh/2]
    float* out = (float*)d_out;                    // [B,T,C] f32

    const int M  = B_ * T_;        // 4096
    const int N1 = 3 * C_;         // 3072
    const int KD = C_;             // 1024

    // Workspace carve-up (256B aligned slabs)
    char* base = (char*)d_ws;
    size_t off = 0;
    auto take = [&](size_t bytes) {
        char* p = base + off;
        off = (off + bytes + 255) & ~(size_t)255;
        return p;
    };
    unsigned short* xb   = (unsigned short*)take((size_t)M * C_ * 2);
    unsigned short* waT  = (unsigned short*)take((size_t)N1 * KD * 2);
    unsigned short* wpT  = (unsigned short*)take((size_t)C_ * C_ * 2);
    float*          qkvf = (float*)take((size_t)M * N1 * 4);
    unsigned short* Qb   = (unsigned short*)take((size_t)M * C_ * 2);
    unsigned short* Kb   = (unsigned short*)take((size_t)M * C_ * 2);
    unsigned short* VTb  = (unsigned short*)take((size_t)M * C_ * 2);
    unsigned short* Yb   = (unsigned short*)take((size_t)M * C_ * 2);
    (void)ws_size; (void)in_sizes; (void)n_in; (void)out_size;

    // 1) convert x to bf16
    {
        int n = M * C_;
        cvt_bf16_kernel<<<(n + 255) / 256, 256, 0, stream>>>(x, xb, n);
    }
    // 2) pack transposed bf16 weights
    {
        int n = KD * N1;
        transpose_bf16_kernel<<<(n + 255) / 256, 256, 0, stream>>>(w_attn, waT, KD, N1);
        n = C_ * C_;
        transpose_bf16_kernel<<<(n + 255) / 256, 256, 0, stream>>>(w_proj, wpT, C_, C_);
    }
    // 3) QKV GEMM: qkvf[M,3C] = xb @ w_attn  (WMMA bf16 -> f32)
    {
        dim3 grid(N1 / 128, M / 64);
        gemm_bf16_wmma<<<grid, 256, 0, stream>>>(xb, waT, qkvf, M, N1, KD);
    }
    // 4) RoPE + head re-layout (Q,K roped; V transposed)
    {
        int n = B_ * T_ * H_ * HD2;
        rope_scatter_kernel<<<(n + 255) / 256, 256, 0, stream>>>(
            qkvf, fcos, fsin, Qb, Kb, VTb);
    }
    // 5) Flash attention (per-wave 16-query tile, online softmax, WMMA, TDM)
    {
        dim3 grid(T_ / 16, H_, B_);
        flash_attn_kernel<<<grid, 32, 0, stream>>>(Qb, Kb, VTb, Yb);
    }
    // 6) Output projection: out[M,C] = Yb @ w_proj
    {
        dim3 grid(C_ / 128, M / 64);
        gemm_bf16_wmma<<<grid, 256, 0, stream>>>(Yb, wpT, out, M, C_, C_);
    }
}
